// MaterializedTransformerBlock_61727269978151
// MI455X (gfx1250) — compile-verified, matrix operand order checked by reading, not run on hardware
//
#include <hip/hip_runtime.h>
#include <hip/hip_bf16.h>

// ---------------- problem constants ----------------
#define Bq  2
#define Sq  256
#define CNq 256
#define CEq 128
#define Hq  8
#define Aq  32
#define Vq  32
#define AHq 256
#define VHq 256

// ---------------- f16 weight region offsets (in halves) ----------------
#define OFF_WM1H 0
#define OFF_WM2H 65536
#define OFF_WQH  131072
#define OFF_WKNH 196608
#define OFF_WVNH 262144
#define OFF_WOH  327680
#define OFF_WE1H 393216
#define OFF_WE2H 409600
#define OFF_WEEH 425984
#define OFF_WENH 442368
#define H_TOTAL  475136           // halves
#define H_BYTES  (H_TOTAL * 2)    // 950272, 16B aligned

#define EJ_LD 136                 // padded LDS leading dim (halves), mult of 8

typedef __attribute__((ext_vector_type(16))) _Float16 v16h;
typedef __attribute__((ext_vector_type(8)))  _Float16 v8h;
typedef __attribute__((ext_vector_type(8)))  float    v8f;
typedef __attribute__((ext_vector_type(4)))  unsigned int u32x4;
typedef __attribute__((ext_vector_type(8)))  int i32x8;
typedef __attribute__((ext_vector_type(4)))  int i32x4;

// Load a 16x32 f16 operand tile (A: rows=M, or B: rows=N from transposed
// storage). Per-lane: row = row0 + (lane&15); two contiguous 8-half chunks at
// K = k0 + (lane>>4)*8 and K+16. Matches CDNA5 16-bit A/B VGPR layout.
__device__ __forceinline__ v16h ld_tile(const _Float16* __restrict__ base,
                                        int ld, int row0, int k0, int lane) {
  const _Float16* p = base + (row0 + (lane & 15)) * ld + k0 + ((lane >> 4) << 3);
  v8h lo = *(const v8h*)(p);
  v8h hi = *(const v8h*)(p + 16);
  v16h out;
#pragma unroll
  for (int e = 0; e < 8; ++e) { out[e] = lo[e]; out[e + 8] = hi[e]; }
  return out;
}

__device__ __forceinline__ v8f wmma16(v16h a, v16h b, v8f c) {
  return __builtin_amdgcn_wmma_f32_16x16x32_f16(false, a, false, b, (short)0, c,
                                                false, false);
}

// ---------------- prep: fp32 weights -> f16 panels in ws ----------------
__global__ void k_prep(const float* __restrict__ Wq, const float* __restrict__ Wk,
                       const float* __restrict__ Wv, const float* __restrict__ Wo,
                       const float* __restrict__ We, const float* __restrict__ Wm1,
                       const float* __restrict__ Wm2, const float* __restrict__ We1,
                       const float* __restrict__ We2, _Float16* __restrict__ H) {
  int t = blockIdx.x * 256 + threadIdx.x;
  if (t >= H_TOTAL) return;
  float v;
  if (t < OFF_WM2H)        v = Wm1[t];
  else if (t < OFF_WQH)    v = Wm2[t - OFF_WM2H];
  else if (t < OFF_WKNH)   v = Wq[t - OFF_WQH];
  else if (t < OFF_WVNH) { int e = t - OFF_WKNH; v = Wk[(e >> 8) * 384 + (e & 255)]; }
  else if (t < OFF_WOH)  { int e = t - OFF_WVNH; v = Wv[(e >> 8) * 384 + (e & 255)]; }
  else if (t < OFF_WE1H)   v = Wo[t - OFF_WOH];
  else if (t < OFF_WE2H)   v = We1[t - OFF_WE1H];
  else if (t < OFF_WEEH)   v = We2[t - OFF_WE2H];
  else if (t < OFF_WENH) { int e = t - OFF_WEEH; v = We[(e >> 7) * 384 + 256 + (e & 127)]; }
  else                   { int e = t - OFF_WENH; v = We[(e >> 8) * 384 + (e & 255)]; }
  H[t] = (_Float16)v;
}

// ---------------- node MLP (ReZero) -> nodes' (fp32 + f16[pos][chan]) ------
#define SBLK 64
__global__ void k_node_mlp(const float* __restrict__ nodes,
                           const float* __restrict__ bm1, const float* __restrict__ bm2,
                           const float* __restrict__ an0,
                           const _Float16* __restrict__ Hws,
                           float* __restrict__ nodesP, _Float16* __restrict__ NPjc) {
  extern __shared__ char smem[];
  _Float16* Njc = (_Float16*)smem;         // [SBLK][256]
  _Float16* H1  = Njc + SBLK * 256;        // [SBLK][256]
  int b  = blockIdx.x >> 2;
  int s0 = (blockIdx.x & 3) * SBLK;
  int tid = threadIdx.x, lane = tid & 31, wv = tid >> 5;
  const _Float16* Wm1h = Hws + OFF_WM1H;
  const _Float16* Wm2h = Hws + OFF_WM2H;

  for (int e = tid; e < CNq * SBLK; e += 256) {
    int c = e >> 6, sl = e & 63;
    Njc[sl * 256 + c] = (_Float16)nodes[(b * CNq + c) * Sq + s0 + sl];
  }
  __syncthreads();

  // G1 = relu(Wm1 @ N + bm1) -> H1[pos][chan]
  for (int tt = wv; tt < 64; tt += 8) {
    int m0 = (tt >> 2) * 16, n0 = (tt & 3) * 16;
    v8f acc = {};
#pragma unroll
    for (int k0 = 0; k0 < 256; k0 += 32)
      acc = wmma16(ld_tile(Wm1h, 256, m0, k0, lane),
                   ld_tile(Njc, 256, n0, k0, lane), acc);
    int n = n0 + (lane & 15), mh = m0 + ((lane >> 4) << 3);
#pragma unroll
    for (int r = 0; r < 8; ++r) {
      int o = mh + r;
      float v = acc[r] + bm1[o];
      H1[n * 256 + o] = (_Float16)(v > 0.f ? v : 0.f);
    }
  }
  __syncthreads();

  // nodes' = nodes + an0 * relu(Wm2 @ G1 + bm2)
  for (int tt = wv; tt < 64; tt += 8) {
    int m0 = (tt >> 2) * 16, n0 = (tt & 3) * 16;
    v8f acc = {};
#pragma unroll
    for (int k0 = 0; k0 < 256; k0 += 32)
      acc = wmma16(ld_tile(Wm2h, 256, m0, k0, lane),
                   ld_tile(H1, 256, n0, k0, lane), acc);
    int n = n0 + (lane & 15), mh = m0 + ((lane >> 4) << 3);
#pragma unroll
    for (int r = 0; r < 8; ++r) {
      int o = mh + r, s = s0 + n;
      float g2 = acc[r] + bm2[o];
      g2 = g2 > 0.f ? g2 : 0.f;
      float np = nodes[(b * CNq + o) * Sq + s] + an0[o] * g2;
      nodesP[(b * CNq + o) * Sq + s] = np;
      NPjc[(b * Sq + s) * CNq + o]   = (_Float16)np;
    }
  }
}

// ---------------- q / kn / vn = W @ nodes' (3x 256x256x256) ----------------
__global__ void k_node_proj(const _Float16* __restrict__ Hws,
                            const _Float16* __restrict__ NPjc,
                            float* __restrict__ qb, float* __restrict__ knb,
                            float* __restrict__ vnb) {
  int id = blockIdx.x;
  int mt = id & 15; id >>= 4;
  int mat = id % 3, b = id / 3;
  const _Float16* W = Hws + (mat == 0 ? OFF_WQH : (mat == 1 ? OFF_WKNH : OFF_WVNH));
  float* Out = (mat == 0 ? qb : (mat == 1 ? knb : vnb));
  const _Float16* Bt = NPjc + b * Sq * CNq;
  int lane = threadIdx.x & 31, wv = threadIdx.x >> 5;
  for (int nt = wv; nt < 16; nt += 8) {
    v8f acc = {};
#pragma unroll
    for (int k0 = 0; k0 < 256; k0 += 32)
      acc = wmma16(ld_tile(W, 256, mt * 16, k0, lane),
                   ld_tile(Bt, 256, nt * 16, k0, lane), acc);
    int n = nt * 16 + (lane & 15), mh = mt * 16 + ((lane >> 4) << 3);
#pragma unroll
    for (int r = 0; r < 8; ++r)
      Out[(b * 256 + mh + r) * Sq + n] = acc[r];
  }
}

// ---------------- fused per-(b,i) row kernel ----------------
// TDM async-loads the 128x256 f32 edge slab to LDS while the wave computes the
// q-fold terms that don't depend on it; then edge MLP -> E' -> sim -> softmax
// -> nf -> partial edge output, edge MLP and We_e @ E' via WMMA, the rest scalar.
#define LDS_EJC   (256 * EJ_LD * 2)                          // 69632
#define LDS_STAGE (CEq * Sq * 4)                             // 131072
#define LDS_SMALL ((256 + 1024 + 2048 + 1024 + 8 + 8) * 4)   // 17472
#define LDS_ROW   (LDS_EJC + LDS_STAGE + LDS_SMALL)          // 218176

__global__ void k_row(const float* __restrict__ edges, const float* __restrict__ Wk,
                      const float* __restrict__ Wv,
                      const float* __restrict__ be1, const float* __restrict__ be2,
                      const float* __restrict__ ae0, const float* __restrict__ ae1,
                      const unsigned char* __restrict__ mask,
                      const _Float16* __restrict__ Hws,
                      const float* __restrict__ qb, const float* __restrict__ knb,
                      const float* __restrict__ vnb,
                      float* __restrict__ nfb, _Float16* __restrict__ NFjc,
                      float* __restrict__ edges_out) {
  extern __shared__ char smem[];
  _Float16* Ejc = (_Float16*)smem;                        // [256][EJ_LD] f16
  float*    stage = (float*)(smem + LDS_EJC);             // [128][256] f32 (TDM)
  _Float16* Gjc = (_Float16*)(smem + LDS_EJC);            // aliases stage (later)
  float* qi   = (float*)(smem + LDS_EJC + LDS_STAGE);     // [256]
  float* wbuf = qi + 256;                                 // [8*128]
  float* simb = wbuf + 8 * 128;                           // [8*256]
  float* zbuf = simb + 8 * 256;                           // [128*8]
  float* chh  = zbuf + 128 * 8;                           // [8]
  float* ssum = chh + 8;                                  // [8]

  int b = blockIdx.x >> 8, i = blockIdx.x & 255;
  int tid = threadIdx.x, lane = tid & 31, wv = tid >> 5;
  const _Float16* We1h = Hws + OFF_WE1H;
  const _Float16* We2h = Hws + OFF_WE2H;
  const _Float16* Weeh = Hws + OFF_WEEH;

  // 0. kick off the TDM for the edge slab (one DMA per workgroup, wave 0)
  qi[tid] = qb[(b * AHq + tid) * Sq + i];
  if (wv == 0) {
    unsigned lds_off = (unsigned)(unsigned long long)stage;   // LDS aperture: low 32b
    unsigned long long ga =
        (unsigned long long)(edges + ((size_t)(b * CEq) * Sq + i) * Sq);
    u32x4 g0;
    g0[0] = 1u;                                          // count=1 (valid user D#)
    g0[1] = lds_off;                                     // lds_addr
    g0[2] = (unsigned)(ga & 0xFFFFFFFFu);                // global_addr[31:0]
    g0[3] = (unsigned)((ga >> 32) & 0x01FFFFFFu) | (2u << 30);  // addr[56:32]|type=2
    i32x8 g1;
    g1[0] = 0x00020000;          // workgroup_mask=0 (not a cluster), data_size=4B
    g1[1] = (int)(256u << 16);   // tensor_dim0 = 256   (bits 79:48, low16 here)
    g1[2] = (int)(128u << 16);   // tensor_dim1 = 128   (bits 111:80, low16 here)
    g1[3] = (int)(256u << 16);   // tile_dim0 = 256     (bits 127:112)
    g1[4] = 128;                 // tile_dim1 = 128     (bits 143:128)
    g1[5] = Sq * Sq;             // tensor_dim0_stride = 65536 elements
    g1[6] = 0;
    g1[7] = 0;
    i32x4 zz = {0, 0, 0, 0};
#if defined(__clang_major__) && (__clang_major__ >= 23)
    i32x8 zz8 = {0, 0, 0, 0, 0, 0, 0, 0};
    __builtin_amdgcn_tensor_load_to_lds(g0, g1, zz, zz, zz8, 0);
#else
    __builtin_amdgcn_tensor_load_to_lds(g0, g1, zz, zz, 0);
#endif
  }
  __syncthreads();   // qi visible (TDM still in flight)

  // 1a. per-head constant c_h = sum_a q[ha,i]*kn[ha,i]  (overlaps with DMA)
  {
    float p = qi[tid] * knb[(b * AHq + tid) * Sq + i];
#pragma unroll
    for (int m = 16; m >= 1; m >>= 1) p += __shfl_xor(p, m, 32);
    if (lane == 0) chh[wv] = p;
  }
  // 1b. w[h][c] = sum_a q[ha,i] * Wk_e[ha,c]            (overlaps with DMA)
  for (int e = tid; e < 1024; e += 256) {
    int h = e >> 7, c = e & 127;
    float acc = 0.f;
#pragma unroll 4
    for (int a = 0; a < 32; ++a)
      acc += qi[h * 32 + a] * Wk[(h * 32 + a) * 384 + 256 + c];
    wbuf[e] = acc;
  }

  // 1c. wait for the DMA, then convert stage f32 -> Ejc f16 [j][ce]
  if (wv == 0) __builtin_amdgcn_s_wait_tensorcnt(0);
  __syncthreads();
  for (int ce = 0; ce < CEq; ++ce)
    Ejc[tid * EJ_LD + ce] = (_Float16)stage[ce * Sq + tid];
  __syncthreads();

  // 2. G1 = relu(We1 @ E + be1)   (Gjc aliases the dead stage region)
  for (int tt = wv; tt < 128; tt += 8) {
    int m0 = (tt >> 4) * 16, n0 = (tt & 15) * 16;
    v8f acc = {};
#pragma unroll
    for (int k0 = 0; k0 < 128; k0 += 32)
      acc = wmma16(ld_tile(We1h, 128, m0, k0, lane),
                   ld_tile(Ejc, EJ_LD, n0, k0, lane), acc);
    int n = n0 + (lane & 15), mh = m0 + ((lane >> 4) << 3);
#pragma unroll
    for (int r = 0; r < 8; ++r) {
      int o = mh + r;
      float v = acc[r] + be1[o];
      Gjc[n * EJ_LD + o] = (_Float16)(v > 0.f ? v : 0.f);
    }
  }
  __syncthreads();

  // 3. E' = E + ae0 * relu(We2 @ G1 + be2)  (in place in Ejc)
  for (int tt = wv; tt < 128; tt += 8) {
    int m0 = (tt >> 4) * 16, n0 = (tt & 15) * 16;
    v8f acc = {};
#pragma unroll
    for (int k0 = 0; k0 < 128; k0 += 32)
      acc = wmma16(ld_tile(We2h, 128, m0, k0, lane),
                   ld_tile(Gjc, EJ_LD, n0, k0, lane), acc);
    int n = n0 + (lane & 15), mh = m0 + ((lane >> 4) << 3);
#pragma unroll
    for (int r = 0; r < 8; ++r) {
      int o = mh + r;
      float g2 = acc[r] + be2[o];
      g2 = g2 > 0.f ? g2 : 0.f;
      int idx = n * EJ_LD + o;
      Ejc[idx] = (_Float16)((float)Ejc[idx] + ae0[o] * g2);
    }
  }
  __syncthreads();

  // 4. sim[h][j] = rsA * (c_h + w_h . E'[j]) ; mask_j -> -1e9
  const float rsA = 0.17677669529663687f;
  for (int e = tid; e < 2048; e += 256) {
    int h = e >> 8, j = e & 255;
    float acc = chh[h];
    const float* wr = wbuf + h * 128;
    const _Float16* er = Ejc + j * EJ_LD;
#pragma unroll 4
    for (int c = 0; c < 128; ++c) acc += wr[c] * (float)er[c];
    acc *= rsA;
    if (!mask[b * Sq + j]) acc = -1e9f;
    simb[e] = acc;
  }
  __syncthreads();

  // 5. masked softmax per head (wave wv == head), then *mm, track row sum
  {
    int h = wv;
    float vloc[8], mx = -1e30f;
#pragma unroll
    for (int r = 0; r < 8; ++r) {
      vloc[r] = simb[h * 256 + lane + r * 32];
      mx = fmaxf(mx, vloc[r]);
    }
#pragma unroll
    for (int m = 16; m >= 1; m >>= 1) mx = fmaxf(mx, __shfl_xor(mx, m, 32));
    float se = 0.f;
#pragma unroll
    for (int r = 0; r < 8; ++r) { vloc[r] = __expf(vloc[r] - mx); se += vloc[r]; }
#pragma unroll
    for (int m = 16; m >= 1; m >>= 1) se += __shfl_xor(se, m, 32);
    float inv = 1.0f / se;
    int mi = mask[b * Sq + i] ? 1 : 0;
    float ss = 0.f;
#pragma unroll
    for (int r = 0; r < 8; ++r) {
      int j = lane + r * 32;
      float sv = vloc[r] * inv;
      sv *= (mi && mask[b * Sq + j]) ? 1.0f : 0.0f;
      simb[h * 256 + j] = sv;
      ss += sv;
    }
#pragma unroll
    for (int m = 16; m >= 1; m >>= 1) ss += __shfl_xor(ss, m, 32);
    if (lane == 0) ssum[h] = ss;
  }
  __syncthreads();

  // 6. z[c][h] = sum_j E'[j][c] * sim[h][j]
  for (int e = tid; e < 1024; e += 256) {
    int c = e >> 3, h = e & 7;
    const float* sr = simb + h * 256;
    float acc = 0.f;
#pragma unroll 4
    for (int j = 0; j < 256; ++j) acc += sr[j] * (float)Ejc[j * EJ_LD + c];
    zbuf[e] = acc;
  }
  __syncthreads();

  // 7. nf[hv] = vn[hv,i]*ssum_h + Wv_e[hv,:] . z[:,h]
  {
    int hv = tid, h = hv >> 5;
    float acc = vnb[(b * VHq + hv) * Sq + i] * ssum[h];
#pragma unroll 4
    for (int c = 0; c < 128; ++c)
      acc += Wv[hv * 384 + 256 + c] * zbuf[c * 8 + h];
    nfb[(b * VHq + hv) * Sq + i] = acc;
    NFjc[(b * Sq + i) * VHq + hv] = (_Float16)acc;
  }

  // 8. partial edge out = E' + ae1 * (We_e @ E')
  for (int tt = wv; tt < 128; tt += 8) {
    int m0 = (tt >> 4) * 16, n0 = (tt & 15) * 16;
    v8f acc = {};
#pragma unroll
    for (int k0 = 0; k0 < 128; k0 += 32)
      acc = wmma16(ld_tile(Weeh, 128, m0, k0, lane),
                   ld_tile(Ejc, EJ_LD, n0, k0, lane), acc);
    int n = n0 + (lane & 15), mh = m0 + ((lane >> 4) << 3);
#pragma unroll
    for (int r = 0; r < 8; ++r) {
      int ce = mh + r;
      float part = (float)Ejc[n * EJ_LD + ce] + ae1[ce] * acc[r];
      edges_out[((size_t)(b * CEq + ce) * Sq + i) * Sq + n] = part;
    }
  }
}

// ---------------- node output + u = We_n @ nf ----------------
__global__ void k_node_out(const _Float16* __restrict__ Hws,
                           const _Float16* __restrict__ NFjc,
                           const float* __restrict__ nodesP,
                           const float* __restrict__ an1,
                           const unsigned char* __restrict__ mask,
                           float* __restrict__ u, float* __restrict__ nodes_out) {
  int id = blockIdx.x;
  int b = id / 24, r24 = id % 24;
  int lane = threadIdx.x & 31, wv = threadIdx.x >> 5;
  const _Float16* Bt = NFjc + b * Sq * VHq;
  if (r24 < 16) {  // node_att = Wo @ nf ; nodes_out = (nodes' + an1*na)*mask_i
    int m0 = r24 * 16;
    const _Float16* W = Hws + OFF_WOH;
    for (int nt = wv; nt < 16; nt += 8) {
      v8f acc = {};
#pragma unroll
      for (int k0 = 0; k0 < 256; k0 += 32)
        acc = wmma16(ld_tile(W, 256, m0, k0, lane),
                     ld_tile(Bt, 256, nt * 16, k0, lane), acc);
      int n = nt * 16 + (lane & 15), mh = m0 + ((lane >> 4) << 3);
#pragma unroll
      for (int r = 0; r < 8; ++r) {
        int o = mh + r, s = n;
        float val = (nodesP[(b * CNq + o) * Sq + s] + an1[o] * acc[r]) *
                    (mask[b * Sq + s] ? 1.0f : 0.0f);
        nodes_out[(b * CNq + o) * Sq + s] = val;
      }
    }
  } else {         // u = We_n @ nf  (128 x 256)
    int m0 = (r24 - 16) * 16;
    const _Float16* W = Hws + OFF_WENH;
    for (int nt = wv; nt < 16; nt += 8) {
      v8f acc = {};
#pragma unroll
      for (int k0 = 0; k0 < 256; k0 += 32)
        acc = wmma16(ld_tile(W, 256, m0, k0, lane),
                     ld_tile(Bt, 256, nt * 16, k0, lane), acc);
      int n = nt * 16 + (lane & 15), mh = m0 + ((lane >> 4) << 3);
#pragma unroll
      for (int r = 0; r < 8; ++r)
        u[(b * CEq + mh + r) * Sq + n] = acc[r];
    }
  }
}

// ------- finalize edges: += ae1*0.5*(u_i+u_j), then mask mm, in place -------
__global__ void k_edge_final(const float* __restrict__ u,
                             const float* __restrict__ ae1,
                             const unsigned char* __restrict__ mask,
                             float* __restrict__ eo) {
  size_t g = (size_t)blockIdx.x * blockDim.x + threadIdx.x;
  size_t base = g * 4;
  int j0 = (int)(base & 255);
  size_t row = base >> 8;
  int i = (int)(row & 255);
  size_t bc = row >> 8;
  int ce = (int)(bc & 127);
  int b = (int)(bc >> 7);
  const float* ur = u + (size_t)(b * CEq + ce) * Sq;
  float ui = ur[i];
  float a = 0.5f * ae1[ce];
  int mi = mask[b * Sq + i];
  float4 pv = *(const float4*)(eo + base);
  float* p = (float*)&pv;
  float m0 = (mi && mask[b * Sq + j0 + 0]) ? 1.0f : 0.0f;
  float m1 = (mi && mask[b * Sq + j0 + 1]) ? 1.0f : 0.0f;
  float m2 = (mi && mask[b * Sq + j0 + 2]) ? 1.0f : 0.0f;
  float m3 = (mi && mask[b * Sq + j0 + 3]) ? 1.0f : 0.0f;
  float o0 = (p[0] + a * (ui + ur[j0 + 0])) * m0;
  float o1 = (p[1] + a * (ui + ur[j0 + 1])) * m1;
  float o2 = (p[2] + a * (ui + ur[j0 + 2])) * m2;
  float o3 = (p[3] + a * (ui + ur[j0 + 3])) * m3;
  *(float4*)(eo + base) = make_float4(o0, o1, o2, o3);
}

// ---------------- host launcher ----------------
extern "C" void kernel_launch(void* const* d_in, const int* in_sizes, int n_in,
                              void* d_out, int out_size, void* d_ws, size_t ws_size,
                              hipStream_t stream) {
  (void)in_sizes; (void)n_in; (void)out_size; (void)ws_size;
  const float* nodes = (const float*)d_in[0];
  const float* edges = (const float*)d_in[1];
  const unsigned char* mask = (const unsigned char*)d_in[2];
  const float* Wq  = (const float*)d_in[3];
  const float* Wk  = (const float*)d_in[4];
  const float* Wv  = (const float*)d_in[5];
  const float* Wo  = (const float*)d_in[6];
  const float* We  = (const float*)d_in[7];
  const float* Wm1 = (const float*)d_in[8];
  const float* bm1 = (const float*)d_in[9];
  const float* Wm2 = (const float*)d_in[10];
  const float* bm2 = (const float*)d_in[11];
  const float* We1 = (const float*)d_in[12];
  const float* be1 = (const float*)d_in[13];
  const float* We2 = (const float*)d_in[14];
  const float* be2 = (const float*)d_in[15];
  const float* an0 = (const float*)d_in[16];
  const float* an1 = (const float*)d_in[17];
  const float* ae0 = (const float*)d_in[18];
  const float* ae1 = (const float*)d_in[19];

  // workspace layout
  char* ws = (char*)d_ws;
  _Float16* Hws = (_Float16*)ws;
  float* nodesP = (float*)(ws + H_BYTES);
  float* qb  = nodesP + Bq * CNq * Sq;      // 131072 each
  float* knb = qb  + Bq * AHq * Sq;
  float* vnb = knb + Bq * AHq * Sq;
  float* nfb = vnb + Bq * VHq * Sq;
  float* ub  = nfb + Bq * VHq * Sq;         // 65536
  _Float16* NPjc = (_Float16*)(ub + Bq * CEq * Sq);
  _Float16* NFjc = NPjc + Bq * Sq * CNq;

  float* nodes_out = (float*)d_out;
  float* edges_out = nodes_out + Bq * CNq * Sq;

  const size_t lds_mlp = (size_t)2 * SBLK * 256 * sizeof(_Float16);  // 64 KB
  const size_t lds_row = (size_t)LDS_ROW;                            // ~213 KB
  hipFuncSetAttribute((const void*)k_node_mlp,
                      hipFuncAttributeMaxDynamicSharedMemorySize, (int)lds_mlp);
  hipFuncSetAttribute((const void*)k_row,
                      hipFuncAttributeMaxDynamicSharedMemorySize, (int)lds_row);

  k_prep<<<(H_TOTAL + 255) / 256, 256, 0, stream>>>(Wq, Wk, Wv, Wo, We, Wm1, Wm2,
                                                    We1, We2, Hws);
  k_node_mlp<<<Bq * (Sq / SBLK), 256, lds_mlp, stream>>>(nodes, bm1, bm2, an0,
                                                         Hws, nodesP, NPjc);
  k_node_proj<<<Bq * 3 * 16, 256, 0, stream>>>(Hws, NPjc, qb, knb, vnb);
  k_row<<<Bq * Sq, 256, lds_row, stream>>>(edges, Wk, Wv, be1, be2, ae0, ae1,
                                           mask, Hws, qb, knb, vnb, nfb, NFjc,
                                           edges_out);
  k_node_out<<<Bq * 24, 256, 0, stream>>>(Hws, NFjc, nodesP, an1, mask, ub,
                                          nodes_out);
  k_edge_final<<<(Bq * CEq * Sq * Sq) / 4 / 256, 256, 0, stream>>>(ub, ae1, mask,
                                                                   edges_out);
}
